// UnifiedMMSEModule_43095701848146
// MI455X (gfx1250) — matrix operand order repaired and checked by reading, not compiled, not run on hardware
//
#include <hip/hip_runtime.h>

// ---------------------------------------------------------------------------
// MMSE fused pipeline for gfx1250 (MI455X, wave32, WMMA).
//   prep:  f32 weights -> bf16, N-major, block-diag packed into d_ws
//   fused: X[8192,4096] -> (bf16 WMMA) -> H1[.,128] -> H2[.,128] -> OUT[.,288]
//          -> per-row complex L*L^H for C and R heads -> d_out
// ---------------------------------------------------------------------------

typedef __attribute__((ext_vector_type(16))) __bf16 v16bf;
typedef __attribute__((ext_vector_type(8)))  float  v8f;

struct alignas(16) U128 { unsigned int x, y, z, w; };
struct alignas(16) F4   { float x, y, z, w; };

union Frag  { U128 q[2]; v16bf v; };
union Pack8 { __bf16 h[8]; U128 u; };

__device__ __forceinline__ v8f v8f_zero() {
    v8f z;
#pragma unroll
    for (int j = 0; j < 8; ++j) z[j] = 0.f;
    return z;
}

// ---------------------------------------------------------------------------
// Workspace layout (bytes):
//   Wt1: bf16 [128][4096]  @ 0          (1,048,576)   n-major: n<64 -> WC1 col, else WR1
//   Wt2: bf16 [128][128]   @ 1,048,576  (32,768)      block-diag(WC2, WR2), n-major
//   Wt3: bf16 [288][128]   @ 1,081,344  (73,728)      block-diag(WC3, WR3), n-major
//   b1 : f32  [128]        @ 1,155,072
//   b2 : f32  [128]        @ 1,155,584
//   b3 : f32  [288]        @ 1,156,096   (total 1,157,248 B)
// ---------------------------------------------------------------------------
#define OFF_WT2 1048576
#define OFF_WT3 1081344
#define OFF_B1  1155072
#define OFF_B2  1155584
#define OFF_B3  1156096

__global__ void mmse_prep(const float* __restrict__ WC1, const float* __restrict__ WR1,
                          const float* __restrict__ WC2, const float* __restrict__ WR2,
                          const float* __restrict__ WC3, const float* __restrict__ WR3,
                          const float* __restrict__ bC1, const float* __restrict__ bR1,
                          const float* __restrict__ bC2, const float* __restrict__ bR2,
                          const float* __restrict__ bC3, const float* __restrict__ bR3,
                          unsigned char* __restrict__ ws) {
    __bf16* Wt1 = (__bf16*)ws;
    __bf16* Wt2 = (__bf16*)(ws + OFF_WT2);
    __bf16* Wt3 = (__bf16*)(ws + OFF_WT3);
    float* b1 = (float*)(ws + OFF_B1);
    float* b2 = (float*)(ws + OFF_B2);
    float* b3 = (float*)(ws + OFF_B3);

    const int NW1 = 128 * 4096, NW2 = 128 * 128, NW3 = 288 * 128;
    const int total = NW1 + NW2 + NW3 + 544;
    const int stride = gridDim.x * blockDim.x;
    for (int t = blockIdx.x * blockDim.x + threadIdx.x; t < total; t += stride) {
        if (t < NW1) {
            int n = t / 4096, k = t % 4096;
            float v = (n < 64) ? WC1[k * 64 + n] : WR1[k * 64 + (n - 64)];
            Wt1[t] = (__bf16)v;
        } else if (t < NW1 + NW2) {
            int u = t - NW1, n = u / 128, k = u % 128;
            float v = 0.f;
            if (n < 64)  { if (k < 64)  v = WC2[k * 64 + n]; }
            else         { if (k >= 64) v = WR2[(k - 64) * 64 + (n - 64)]; }
            Wt2[u] = (__bf16)v;
        } else if (t < NW1 + NW2 + NW3) {
            int u = t - NW1 - NW2, n = u / 128, k = u % 128;
            float v = 0.f;
            if (n < 144) { if (k < 64)  v = WC3[k * 144 + n]; }
            else         { if (k >= 64) v = WR3[(k - 64) * 144 + (n - 144)]; }
            Wt3[u] = (__bf16)v;
        } else {
            int u = t - NW1 - NW2 - NW3;
            if (u < 128)       b1[u] = (u < 64) ? bC1[u] : bR1[u - 64];
            else if (u < 256)  { int n = u - 128; b2[n] = (n < 64) ? bC2[n] : bR2[n - 64]; }
            else               { int n = u - 256; b3[n] = (n < 144) ? bC3[n] : bR3[n - 144]; }
        }
    }
}

// ---------------------------------------------------------------------------
// Fused kernel: 256 threads (8 wave32), 32 batch rows per block.
// ---------------------------------------------------------------------------
__global__ __launch_bounds__(256) void mmse_fused(const float* __restrict__ X,
                                                  const unsigned char* __restrict__ ws,
                                                  float* __restrict__ out, int rows) {
    const __bf16* Wt1 = (const __bf16*)ws;
    const __bf16* Wt2 = (const __bf16*)(ws + OFF_WT2);
    const __bf16* Wt3 = (const __bf16*)(ws + OFF_WT3);
    const float* b1 = (const float*)(ws + OFF_B1);
    const float* b2 = (const float*)(ws + OFF_B2);
    const float* b3 = (const float*)(ws + OFF_B3);

    // LDS, aliased across stages (45,056 B total):
    //   [0,8192)       H2s  bf16 [32][128]
    //   [8192,12288)   Xs0  bf16 [32][64]    (stage 1, ping)
    //   [12288,16384)  Xs1  bf16 [32][64]    (stage 1, pong)
    //   [16384,24576)  H1s  bf16 [32][128]   (stages 1-2)
    //   [8192,45056)   Outs f32  [32][288]   (stages 3-4, overlays Xs/H1s)
    __shared__ __align__(16) unsigned char smem[45056];
    __bf16* H2s  = (__bf16*)smem;
    __bf16* Xs0  = (__bf16*)(smem + 8192);
    __bf16* Xs1  = (__bf16*)(smem + 12288);
    __bf16* H1s  = (__bf16*)(smem + 16384);
    float*  Outs = (float*)(smem + 8192);

    const int tid   = threadIdx.x;
    const int wave  = tid >> 5;
    const int lane  = tid & 31;
    const int row16 = lane & 15;
    const int lh    = lane >> 4;       // lane-half
    const int kb    = lh * 8;          // A-fragment K sub-offset (ISA 16-bit A layout)
    const int koff  = lh * 8;          // B-fragment K sub-offset (same layout as A)
    const int tm    = wave & 1;        // M tile (2 x 16 rows)
    const int tn0   = (wave >> 1) * 2; // first of 2 N tiles for this wave
    const int r0    = blockIdx.x * 32;

    // ---------------- Stage 1: H1 = relu(X @ W1' + b1), K = 4096 ----------------
    // Double-buffered 32x64 X chunk; one barrier per iteration.
    v8f acc[2];
    acc[0] = v8f_zero(); acc[1] = v8f_zero();

    const int lr = tid >> 3;                 // 0..31 row of X chunk
    const int lc = (tid & 7) * 8;            // 0..56 col (8 floats per thread)
    const float* xbase = X + (size_t)(r0 + lr) * 4096 + lc;

    // preload chunk 0
    F4 f0 = *(const F4*)(xbase);
    F4 f1 = *(const F4*)(xbase + 4);
    {
        Pack8 p;
        p.h[0] = (__bf16)f0.x; p.h[1] = (__bf16)f0.y; p.h[2] = (__bf16)f0.z; p.h[3] = (__bf16)f0.w;
        p.h[4] = (__bf16)f1.x; p.h[5] = (__bf16)f1.y; p.h[6] = (__bf16)f1.z; p.h[7] = (__bf16)f1.w;
        *(U128*)&Xs0[lr * 64 + lc] = p.u;
    }
    __syncthreads();

    for (int kc = 0; kc < 64; ++kc) {
        const int k0 = kc * 64;
        // issue next chunk's global load before the WMMAs (latency overlap)
        F4 g0, g1;
        if (kc < 63) {
            const float* p = xbase + k0 + 64;
            g0 = *(const F4*)p;
            g1 = *(const F4*)(p + 4);
            if (kc < 62) __builtin_prefetch(xbase + k0 + 128, 0, 0);  // global_prefetch_b8
        }

        const __bf16* cur = (kc & 1) ? Xs1 : Xs0;
        const __bf16* ap  = cur + (tm * 16 + row16) * 64;
#pragma unroll
        for (int s = 0; s < 2; ++s) {        // two K=32 sub-chunks
            Frag a;
            a.q[0] = *(const U128*)(ap + s * 32 + kb);
            a.q[1] = *(const U128*)(ap + s * 32 + 16 + kb);
#pragma unroll
            for (int i = 0; i < 2; ++i) {
                const int col = (tn0 + i) * 16 + row16;
                const __bf16* bp = Wt1 + (size_t)col * 4096 + k0 + s * 32 + koff;
                Frag b;
                b.q[0] = *(const U128*)bp;
                b.q[1] = *(const U128*)(bp + 16);
                acc[i] = __builtin_amdgcn_wmma_f32_16x16x32_bf16(
                    false, a.v, false, b.v, (short)0, acc[i], false, false);
            }
        }

        if (kc < 63) {
            __bf16* nxt = (kc & 1) ? Xs0 : Xs1;
            Pack8 p;
            p.h[0] = (__bf16)g0.x; p.h[1] = (__bf16)g0.y; p.h[2] = (__bf16)g0.z; p.h[3] = (__bf16)g0.w;
            p.h[4] = (__bf16)g1.x; p.h[5] = (__bf16)g1.y; p.h[6] = (__bf16)g1.z; p.h[7] = (__bf16)g1.w;
            *(U128*)&nxt[lr * 64 + lc] = p.u;
        }
        __syncthreads();
    }

    // bias + relu -> H1s (bf16)
#pragma unroll
    for (int i = 0; i < 2; ++i) {
        const int n = (tn0 + i) * 16 + row16;
        const float bias = b1[n];
#pragma unroll
        for (int v = 0; v < 8; ++v) {
            const int m = v + lh * 8;
            H1s[(tm * 16 + m) * 128 + n] = (__bf16)fmaxf(acc[i][v] + bias, 0.f);
        }
    }
    __syncthreads();

    // ---------------- Stage 2: H2 = relu(H1 @ W2' + b2), K = 128 ----------------
    v8f acc2[2];
    acc2[0] = v8f_zero(); acc2[1] = v8f_zero();
#pragma unroll
    for (int kc = 0; kc < 4; ++kc) {
        const int k0 = kc * 32;
        Frag a;
        const __bf16* ap = &H1s[(tm * 16 + row16) * 128 + k0];
        a.q[0] = *(const U128*)(ap + kb);
        a.q[1] = *(const U128*)(ap + 16 + kb);
#pragma unroll
        for (int i = 0; i < 2; ++i) {
            const int col = (tn0 + i) * 16 + row16;
            const __bf16* bp = Wt2 + (size_t)col * 128 + k0 + koff;
            Frag b;
            b.q[0] = *(const U128*)bp;
            b.q[1] = *(const U128*)(bp + 16);
            acc2[i] = __builtin_amdgcn_wmma_f32_16x16x32_bf16(
                false, a.v, false, b.v, (short)0, acc2[i], false, false);
        }
    }
#pragma unroll
    for (int i = 0; i < 2; ++i) {
        const int n = (tn0 + i) * 16 + row16;
        const float bias = b2[n];
#pragma unroll
        for (int v = 0; v < 8; ++v) {
            const int m = v + lh * 8;
            H2s[(tm * 16 + m) * 128 + n] = (__bf16)fmaxf(acc2[i][v] + bias, 0.f);
        }
    }
    __syncthreads();

    // ---------------- Stage 3: OUT = H2 @ W3' + b3 (no relu), 32x288 ------------
    for (int t = wave; t < 36; t += 8) {   // 36 tiles = 2 (M) x 18 (N)
        const int tm3 = t & 1, tn3 = t >> 1;
        v8f acc3 = v8f_zero();
#pragma unroll
        for (int kc = 0; kc < 4; ++kc) {
            const int k0 = kc * 32;
            Frag a;
            const __bf16* ap = &H2s[(tm3 * 16 + row16) * 128 + k0];
            a.q[0] = *(const U128*)(ap + kb);
            a.q[1] = *(const U128*)(ap + 16 + kb);
            Frag b;
            const __bf16* bp = Wt3 + (size_t)(tn3 * 16 + row16) * 128 + k0 + koff;
            b.q[0] = *(const U128*)bp;
            b.q[1] = *(const U128*)(bp + 16);
            acc3 = __builtin_amdgcn_wmma_f32_16x16x32_bf16(
                false, a.v, false, b.v, (short)0, acc3, false, false);
        }
        const int n = tn3 * 16 + row16;
        const float bias = b3[n];
#pragma unroll
        for (int v = 0; v < 8; ++v) {
            const int m = v + lh * 8;
            Outs[(tm3 * 16 + m) * 288 + n] = acc3[v] + bias;
        }
    }
    __syncthreads();

    // ---------------- Stage 4a: softplus on diagonal entries (in place) ---------
    for (int e = tid; e < 32 * 24; e += 256) {
        const int row = e / 24, rem = e % 24;
        const int mat = rem / 12, i = rem % 12;
        const int idx = row * 288 + mat * 144 + (i * (i + 1)) / 2 + i;
        const float x = Outs[idx];
        Outs[idx] = fmaxf(x, 0.f) + log1pf(expf(-fabsf(x)));
    }
    __syncthreads();

    // ---------------- Stage 4b: A = L * L^H, complex, interleaved output --------
    // out layout: C[rows][12][12] complex (re,im) then R[rows][12][12].
    const size_t matStride = (size_t)rows * 144 * 2;
    for (int e = tid; e < 32 * 288; e += 256) {
        const int row = e / 288, rem = e % 288;
        const int mat = rem / 144, pos = rem % 144;
        const int i = pos / 12, j = pos % 12;
        const float* Lb = &Outs[row * 288 + mat * 144];
        float sr = 0.f, si = 0.f;
        const int kmax = (i < j) ? i : j;
        for (int k = 0; k <= kmax; ++k) {
            const float ar = Lb[(i * (i + 1)) / 2 + k];
            const float ai = (k < i) ? Lb[78 + (i * (i - 1)) / 2 + k] : 0.f;
            const float br = Lb[(j * (j + 1)) / 2 + k];
            const float bi = (k < j) ? Lb[78 + (j * (j - 1)) / 2 + k] : 0.f;
            sr += ar * br + ai * bi;          // L[i,:] . conj(L[j,:])
            si += ai * br - ar * bi;
        }
        float* op = out + (size_t)mat * matStride + ((size_t)(r0 + row) * 144 + pos) * 2;
        op[0] = sr;
        op[1] = si;
    }
}

// ---------------------------------------------------------------------------
extern "C" void kernel_launch(void* const* d_in, const int* in_sizes, int n_in,
                              void* d_out, int out_size, void* d_ws, size_t ws_size,
                              hipStream_t stream) {
    const float* X   = (const float*)d_in[0];
    const float* WC1 = (const float*)d_in[1];  const float* bC1 = (const float*)d_in[2];
    const float* WC2 = (const float*)d_in[3];  const float* bC2 = (const float*)d_in[4];
    const float* WC3 = (const float*)d_in[5];  const float* bC3 = (const float*)d_in[6];
    const float* WR1 = (const float*)d_in[7];  const float* bR1 = (const float*)d_in[8];
    const float* WR2 = (const float*)d_in[9];  const float* bR2 = (const float*)d_in[10];
    const float* WR3 = (const float*)d_in[11]; const float* bR3 = (const float*)d_in[12];

    const int rows = in_sizes[0] / 4096;                 // 8192
    unsigned char* ws = (unsigned char*)d_ws;
    float* out = (float*)d_out;

    const int prepTotal = 128 * 4096 + 128 * 128 + 288 * 128 + 544;
    mmse_prep<<<(prepTotal + 255) / 256, 256, 0, stream>>>(
        WC1, WR1, WC2, WR2, WC3, WR3, bC1, bR1, bC2, bR2, bC3, bR3, ws);

    mmse_fused<<<rows / 32, 256, 0, stream>>>(X, ws, out, rows);
}
